// BlockSparseAttentionLayer_58136677318745
// MI455X (gfx1250) — matrix operand order repaired and checked by reading, not compiled, not run on hardware
//
#include <hip/hip_runtime.h>

// ---------------------------------------------------------------------------
// Block-sparse causal flash attention for MI455X (gfx1250, wave32, WMMA).
// B=1, H=8, S=4096, D=128, BLOCK=64, LOCAL_BLOCKS=16, VERT_STRIDE=8.
// One workgroup = one (head, q-block), 128 threads = 4 wave32.
//   - K/V fp32 blocks DMA'd global->LDS (global_load_async_to_lds_b128,
//     ASYNCcnt ping-pong) prefetching the next active sparse block while
//     WMMAs run on the current one.
//   - One cooperative fp32->f16 pass per block builds Ks (row-major) and
//     Vt (transposed) for cheap b128 fragment loads.
//   - sm_scale * log2(e) folded into Q; softmax uses raw v_exp_f32 (exp2).
//   - Matrix math: v_wmma_f32_16x16x32_f16; reductions: ds_swizzle xor-mode.
// ---------------------------------------------------------------------------

#define NH          8
#define SEQL        4096
#define HD          128
#define BLK         64
#define NB          (SEQL / BLK)
#define LOCAL_BLKS  16
#define VSTRIDE     8

typedef __attribute__((ext_vector_type(16))) _Float16 v16h;
typedef __attribute__((ext_vector_type(8)))  float    v8f;

typedef unsigned int u32x4 __attribute__((ext_vector_type(4)));
typedef u32x4 __attribute__((may_alias, aligned(16))) u32x4_a;

// raw hardware exp2 (v_exp_f32); softmax args are <= 0 so no range fixup needed
#if __has_builtin(__builtin_amdgcn_exp2f)
#define EXP2(x) __builtin_amdgcn_exp2f(x)
#else
#define EXP2(x) exp2f(x)
#endif

// ds_swizzle xor-mode: imm = (xor_mask<<10) | 0x1f
#define SWZ_XOR(x, imm) \
    __int_as_float(__builtin_amdgcn_ds_swizzle(__float_as_int(x), (imm)))
#define RED_MAX(x)                         \
    do {                                   \
        x = fmaxf(x, SWZ_XOR(x, 0x041f));  \
        x = fmaxf(x, SWZ_XOR(x, 0x081f));  \
        x = fmaxf(x, SWZ_XOR(x, 0x101f));  \
        x = fmaxf(x, SWZ_XOR(x, 0x201f));  \
    } while (0)
#define RED_SUM(x)                         \
    do {                                   \
        x += SWZ_XOR(x, 0x041f);           \
        x += SWZ_XOR(x, 0x081f);           \
        x += SWZ_XOR(x, 0x101f);           \
        x += SWZ_XOR(x, 0x201f);           \
    } while (0)

// f16 A/P fragment from LDS: elems 0..7 at p, elems 8..15 at p+16 halves.
static __device__ inline v16h frag_a_ld(const _Float16* p) {
    union { u32x4_a q[2]; v16h h; } U;
    U.q[0] = *(const u32x4_a*)(p);
    U.q[1] = *(const u32x4_a*)(p + 16);
    return U.h;
}

// f16 B fragment from LDS: elems 0..7 at p, elems 8..15 at p+16 halves.
static __device__ inline v16h frag_b_ld(const _Float16* p) {
    union { u32x4_a q[2]; v16h h; } U;
    U.q[0] = *(const u32x4_a*)(p);
    U.q[1] = *(const u32x4_a*)(p + 16);
    return U.h;
}

__global__ __launch_bounds__(128)
void bs_attn_kernel(const float* __restrict__ q,
                    const float* __restrict__ k,
                    const float* __restrict__ v,
                    const float* __restrict__ sm_scale,
                    float* __restrict__ out) {
    // 168 KB static LDS (gfx1250: up to 320 KB per workgroup)
    __shared__ __align__(16) float    stageK[2][BLK * HD];  // 64 KB fp32 ping-pong
    __shared__ __align__(16) float    stageV[2][BLK * HD];  // 64 KB fp32 ping-pong
    __shared__ __align__(16) _Float16 Ks[BLK * HD];         // 16 KB f16 row-major
    __shared__ __align__(16) _Float16 Vt[HD * BLK];         // 16 KB f16 transposed [d][k]
    __shared__ __align__(16) _Float16 Ps[4][16 * BLK];      //  8 KB per-wave P strips

    const int qb   = blockIdx.x;
    const int h    = blockIdx.y;
    const int tid  = threadIdx.x;
    const int lane = tid & 31;
    const int wave = tid >> 5;
    const int lo   = lane & 15;
    const int hi   = lane >> 4;
    // fold sm_scale and the exp->exp2 base change into Q
    const float qscale = sm_scale[0] * 1.44269504088896340736f;

    // ---- Q A-fragments straight from global into VGPRs (pre-scaled) ----
    v16h qf[4];
    {
        const float* qrow = q + ((size_t)h * SEQL + (size_t)qb * BLK + wave * 16 + lo) * HD;
#pragma unroll
        for (int c = 0; c < 4; ++c) {
            const float* p = qrow + c * 32 + 8 * hi;
            float4 a0 = *(const float4*)(p);
            float4 a1 = *(const float4*)(p + 4);
            float4 b0 = *(const float4*)(p + 16);
            float4 b1 = *(const float4*)(p + 20);
            v16h r;
            r[0]  = (_Float16)(a0.x * qscale); r[1]  = (_Float16)(a0.y * qscale);
            r[2]  = (_Float16)(a0.z * qscale); r[3]  = (_Float16)(a0.w * qscale);
            r[4]  = (_Float16)(a1.x * qscale); r[5]  = (_Float16)(a1.y * qscale);
            r[6]  = (_Float16)(a1.z * qscale); r[7]  = (_Float16)(a1.w * qscale);
            r[8]  = (_Float16)(b0.x * qscale); r[9]  = (_Float16)(b0.y * qscale);
            r[10] = (_Float16)(b0.z * qscale); r[11] = (_Float16)(b0.w * qscale);
            r[12] = (_Float16)(b1.x * qscale); r[13] = (_Float16)(b1.y * qscale);
            r[14] = (_Float16)(b1.z * qscale); r[15] = (_Float16)(b1.w * qscale);
            qf[c] = r;
        }
    }

    // ---- flash state (all in log2 domain) ----
    v8f acc[8];
#pragma unroll
    for (int dt = 0; dt < 8; ++dt)
#pragma unroll
        for (int r = 0; r < 8; ++r) acc[dt][r] = 0.0f;

    float mrow[8], lrow[8];
#pragma unroll
    for (int r = 0; r < 8; ++r) { mrow[r] = -1.0e30f; lrow[r] = 0.0f; }

    // ---- sparse-block scan (uniform SALU work) ----
    auto next_active = [&](int kb) -> int {
        while (kb <= qb) {
            if ((qb - kb) < LOCAL_BLKS || ((kb + h + 1) % VSTRIDE) == 0) return kb;
            ++kb;
        }
        return -1;
    };

    // ---- async global->LDS prefetch of one k-block (fp32 K + V) ----
    // 16 issues/thread * 2 tensors = 32 async b128 ops per wave per batch.
    auto issue_prefetch = [&](int kb, int buf) {
        const float* Kg = k + ((size_t)h * SEQL + (size_t)kb * BLK) * HD;
        const float* Vg = v + ((size_t)h * SEQL + (size_t)kb * BLK) * HD;
        unsigned ldsK = (unsigned)(uintptr_t)&stageK[buf][0];
        unsigned ldsV = (unsigned)(uintptr_t)&stageV[buf][0];
#pragma unroll
        for (int i = 0; i < 16; ++i) {
            unsigned off = (unsigned)(tid + i * 128) * 16u;  // bytes
            asm volatile("global_load_async_to_lds_b128 %0, %1, %2"
                         :: "v"(ldsK + off), "v"(off), "s"(Kg) : "memory");
            asm volatile("global_load_async_to_lds_b128 %0, %1, %2"
                         :: "v"(ldsV + off), "v"(off), "s"(Vg) : "memory");
        }
    };

    int kb  = next_active(0);   // diagonal block guarantees kb >= 0
    int buf = 0;
    issue_prefetch(kb, 0);

    while (kb >= 0) {
        const int kb2 = next_active(kb + 1);

        __syncthreads();  // everyone done reading the buffer we overwrite next
        if (kb2 >= 0) {
            issue_prefetch(kb2, buf ^ 1);
            asm volatile("s_wait_asynccnt 0x20" ::: "memory");  // older batch landed
        } else {
            asm volatile("s_wait_asynccnt 0x0" ::: "memory");
        }
        __syncthreads();  // all waves' segments of stage[buf] visible

        // ---- cooperative fp32->f16 pass: Ks (row-major) + Vt (transposed) ----
        {
            const float* Kf = &stageK[buf][0];
            const float* Vf = &stageV[buf][0];
#pragma unroll
            for (int i = 0; i < 16; ++i) {
                int idx = tid + i * 128;
                int o   = idx * 4;
                float4 f = ((const float4*)Kf)[idx];
                Ks[o + 0] = (_Float16)f.x; Ks[o + 1] = (_Float16)f.y;
                Ks[o + 2] = (_Float16)f.z; Ks[o + 3] = (_Float16)f.w;
                float4 g = ((const float4*)Vf)[idx];
                int kr = o >> 7;
                int d  = o & (HD - 1);
                Vt[(d + 0) * BLK + kr] = (_Float16)g.x;
                Vt[(d + 1) * BLK + kr] = (_Float16)g.y;
                Vt[(d + 2) * BLK + kr] = (_Float16)g.z;
                Vt[(d + 3) * BLK + kr] = (_Float16)g.w;
            }
        }
        __syncthreads();

        // ---- S = Q K^T : 4 N-tiles x 4 K-steps = 16 WMMA ----
        v8f s[4];
#pragma unroll
        for (int nt = 0; nt < 4; ++nt)
#pragma unroll
            for (int r = 0; r < 8; ++r) s[nt][r] = 0.0f;

#pragma unroll
        for (int nt = 0; nt < 4; ++nt) {
            const _Float16* krow = Ks + (nt * 16 + lo) * HD;
#pragma unroll
            for (int c = 0; c < 4; ++c) {
                v16h bf = frag_b_ld(krow + c * 32 + 8 * hi);
                s[nt] = __builtin_amdgcn_wmma_f32_16x16x32_f16(
                    false, qf[c], false, bf, (short)0, s[nt], false, false);
            }
        }

        // ---- token-level causal mask inside the diagonal block ----
        if (kb == qb) {
#pragma unroll
            for (int nt = 0; nt < 4; ++nt)
#pragma unroll
                for (int r = 0; r < 8; ++r) {
                    int qt = wave * 16 + r + 8 * hi;
                    int kt = nt * 16 + lo;
                    if (kt > qt) s[nt][r] = -1.0e30f;
                }
        }

        // ---- online softmax update (base-2, scale pre-folded into Q) ----
#pragma unroll
        for (int r = 0; r < 8; ++r) {
            float mx = fmaxf(fmaxf(s[0][r], s[1][r]), fmaxf(s[2][r], s[3][r]));
            RED_MAX(mx);
            float mnew = fmaxf(mrow[r], mx);
            float corr = EXP2(mrow[r] - mnew);
            float rs = 0.0f;
#pragma unroll
            for (int nt = 0; nt < 4; ++nt) {
                float p = EXP2(s[nt][r] - mnew);
                s[nt][r] = p;
                rs += p;
            }
            RED_SUM(rs);
            lrow[r] = lrow[r] * corr + rs;
            mrow[r] = mnew;
#pragma unroll
            for (int dt = 0; dt < 8; ++dt) acc[dt][r] *= corr;
        }

        // ---- P: C-layout -> A-layout via per-wave LDS strip ----
        _Float16* pb = &Ps[wave][0];
#pragma unroll
        for (int nt = 0; nt < 4; ++nt)
#pragma unroll
            for (int r = 0; r < 8; ++r)
                pb[(r + 8 * hi) * BLK + nt * 16 + lo] = (_Float16)s[nt][r];
        // same-wave DS RAW handled by compiler-inserted s_wait_dscnt

        const _Float16* prow = pb + lo * BLK;
        v16h pf0 = frag_a_ld(prow + 8 * hi);
        v16h pf1 = frag_a_ld(prow + 32 + 8 * hi);

        // ---- O += P V : 8 D-tiles x 2 K-steps = 16 WMMA ----
#pragma unroll
        for (int dt = 0; dt < 8; ++dt) {
            const _Float16* vrow = Vt + (dt * 16 + lo) * BLK;
            v16h b0 = frag_b_ld(vrow + 8 * hi);
            acc[dt] = __builtin_amdgcn_wmma_f32_16x16x32_f16(
                false, pf0, false, b0, (short)0, acc[dt], false, false);
            v16h b1 = frag_b_ld(vrow + 32 + 8 * hi);
            acc[dt] = __builtin_amdgcn_wmma_f32_16x16x32_f16(
                false, pf1, false, b1, (short)0, acc[dt], false, false);
        }

        kb  = kb2;
        buf ^= 1;
    }

    // ---- normalize and store (fp32) ----
    float* Og = out + ((size_t)h * SEQL + (size_t)qb * BLK) * HD;
#pragma unroll
    for (int r = 0; r < 8; ++r) {
        float inv = 1.0f / lrow[r];
        int row = wave * 16 + r + 8 * hi;
#pragma unroll
        for (int dt = 0; dt < 8; ++dt)
            Og[row * HD + dt * 16 + lo] = acc[dt][r] * inv;
    }
}

extern "C" void kernel_launch(void* const* d_in, const int* in_sizes, int n_in,
                              void* d_out, int out_size, void* d_ws, size_t ws_size,
                              hipStream_t stream) {
    (void)in_sizes; (void)n_in; (void)out_size; (void)d_ws; (void)ws_size;
    const float* q        = (const float*)d_in[0];
    const float* k        = (const float*)d_in[1];
    const float* v        = (const float*)d_in[2];
    const float* sm_scale = (const float*)d_in[3];
    float* out            = (float*)d_out;

    dim3 grid(NB, NH);   // 64 q-blocks x 8 heads
    dim3 block(128);     // 4 wave32
    bs_attn_kernel<<<grid, block, 0, stream>>>(q, k, v, sm_scale, out);
}
